// CausalSelfAttention_51178830299721
// MI455X (gfx1250) — matrix operand order repaired and checked by reading, not compiled, not run on hardware
//
#include <hip/hip_runtime.h>
#include <hip/hip_bf16.h>

typedef __attribute__((ext_vector_type(16))) __bf16       v16bf;
typedef __attribute__((ext_vector_type(8)))  float        v8f;
typedef __attribute__((ext_vector_type(4)))  unsigned int u32x4;
typedef __attribute__((ext_vector_type(4)))  float        f32x4;
typedef __attribute__((ext_vector_type(4)))  __bf16       bf16x4;

union ABFrag { v16bf v; u32x4 q[2]; };
union CFrag  { v8f v; float e[8]; };

#define LOG2E 1.4426950408889634f

#define WMMA_BF16(Afrag, Bfrag, Cfrag) \
  __builtin_amdgcn_wmma_f32_16x16x32_bf16(false, (Afrag).v, false, (Bfrag).v, \
                                          (short)0, (Cfrag).v, false, false)

// Async 16B global->LDS copy (CDNA5): vdst = LDS byte addr, vaddr = 32b byte
// offset, saddr = 64b base. Tracked by ASYNCcnt.
__device__ __forceinline__ void async_b128(unsigned lds, unsigned goff,
                                           unsigned long long base) {
  asm volatile("global_load_async_to_lds_b128 %0, %1, %2"
               :: "v"(lds), "v"(goff), "s"(base) : "memory");
}

// ---------------------------------------------------------------------------
// Phase 0a: fp32 -> bf16 elementwise convert
// ---------------------------------------------------------------------------
__global__ __launch_bounds__(256) void cvt_kernel(
    const float* __restrict__ in, __bf16* __restrict__ out, int n) {
  int i = (blockIdx.x * 256 + threadIdx.x) * 4;
  if (i < n) {
    f32x4 f = *(const f32x4*)(in + i);
    bf16x4 o;
    o.x = (__bf16)f.x; o.y = (__bf16)f.y; o.z = (__bf16)f.z; o.w = (__bf16)f.w;
    *(bf16x4*)(out + i) = o;
  }
}

// Phase 0b: fp32 [K][N] -> bf16 [N][K] (transpose so GEMM B-tiles are pure copies)
__global__ __launch_bounds__(256) void cvtT_kernel(
    const float* __restrict__ in, __bf16* __restrict__ out, int K, int N) {
  int idx = blockIdx.x * 256 + threadIdx.x;
  if (idx < K * N) {
    int n = idx % N, k = idx / N;
    out[(size_t)n * K + k] = (__bf16)in[idx];
  }
}

// ---------------------------------------------------------------------------
// Phase 1: qkv = Xb @ WabT^T + b, routed to Q/K [B,H,T,D] and V^T [B,H,D,T].
// Block = 128 threads (4 waves), tile 128x64, K-step 32, double-buffered
// async LDS staging. Each wave: 2 A-frags x 4 B-frags = 8 WMMAs / K-step.
// ---------------------------------------------------------------------------
__global__ __launch_bounds__(128) void qkv_gemm_kernel(
    const __bf16* __restrict__ Xb, const __bf16* __restrict__ WabT,
    const float* __restrict__ bias,
    __bf16* __restrict__ Qb, __bf16* __restrict__ Kb, __bf16* __restrict__ VbT)
{
  __shared__ __align__(16) __bf16 As[2][128][40];
  __shared__ __align__(16) __bf16 Bs[2][64][40];

  const int row0 = blockIdx.x * 128;
  const int col0 = blockIdx.y * 64;
  const int tid  = threadIdx.x;
  const int lane = tid & 31, wv = tid >> 5;
  const int lm = lane & 15, hf = lane >> 4;
  const unsigned long long abase = (unsigned long long)Xb;
  const unsigned long long bbase = (unsigned long long)WabT;

  auto issue_tile = [&](int k0, int bf) {
#pragma unroll
    for (int i = 0; i < 4; ++i) {                       // A: 512 chunks of 16B
      int ci = tid + i * 128;
      int row = ci >> 2, c8 = (ci & 3) << 3;
      async_b128((unsigned)(size_t)&As[bf][row][c8],
                 (unsigned)(((row0 + row) * 1024 + k0 + c8) * 2), abase);
    }
#pragma unroll
    for (int i = 0; i < 2; ++i) {                       // B: 256 chunks of 16B
      int ci = tid + i * 128;
      int nr = ci >> 2, c8 = (ci & 3) << 3;
      async_b128((unsigned)(size_t)&Bs[bf][nr][c8],
                 (unsigned)(((col0 + nr) * 1024 + k0 + c8) * 2), bbase);
    }
  };

  CFrag acc[2][4];
#pragma unroll
  for (int u = 0; u < 2; ++u)
#pragma unroll
    for (int j = 0; j < 4; ++j)
#pragma unroll
      for (int r = 0; r < 8; ++r) acc[u][j].e[r] = 0.0f;

  issue_tile(0, 0);
  for (int kt = 0; kt < 32; ++kt) {
    const int bf = kt & 1;
    if (kt + 1 < 32) {
      issue_tile((kt + 1) * 32, bf ^ 1);
      asm volatile("s_wait_asynccnt 0x6" ::: "memory");  // tile kt's 6 done
    } else {
      asm volatile("s_wait_asynccnt 0x0" ::: "memory");
    }
    __syncthreads();

    ABFrag a0, a1;
    a0.q[0] = *(const u32x4*)&As[bf][wv * 32 + lm][hf * 8];
    a0.q[1] = *(const u32x4*)&As[bf][wv * 32 + lm][16 + hf * 8];
    a1.q[0] = *(const u32x4*)&As[bf][wv * 32 + 16 + lm][hf * 8];
    a1.q[1] = *(const u32x4*)&As[bf][wv * 32 + 16 + lm][16 + hf * 8];
#pragma unroll
    for (int j = 0; j < 4; ++j) {
      ABFrag b;
      b.q[0] = *(const u32x4*)&Bs[bf][j * 16 + lm][hf * 8];
      b.q[1] = *(const u32x4*)&Bs[bf][j * 16 + lm][16 + hf * 8];
      acc[0][j].v = WMMA_BF16(a0, b, acc[0][j]);
      acc[1][j].v = WMMA_BF16(a1, b, acc[1][j]);
    }
    __syncthreads();
  }

#pragma unroll
  for (int j = 0; j < 4; ++j) {
    int c = col0 + j * 16 + lm;
    float bv = bias[c];
    int seg = c >> 10, cc = c & 1023;
    int hh = cc >> 6, dd = cc & 63;
#pragma unroll
    for (int u = 0; u < 2; ++u)
#pragma unroll
      for (int r = 0; r < 8; ++r) {
        int rr = row0 + wv * 32 + u * 16 + r + hf * 8;
        int bb = rr >> 11, tt = rr & 2047;
        float val = acc[u][j].e[r] + bv;
        if (seg == 0)
          Qb[(((size_t)(bb * 16 + hh) * 2048) + tt) * 64 + dd] = (__bf16)val;
        else if (seg == 1)
          Kb[(((size_t)(bb * 16 + hh) * 2048) + tt) * 64 + dd] = (__bf16)val;
        else
          VbT[(((size_t)(bb * 16 + hh) * 64) + dd) * 2048 + tt] = (__bf16)val;
      }
  }
}

// ---------------------------------------------------------------------------
// Phase 2: flash-style causal attention; K tile and pre-transposed V tile are
// double-buffered async LDS copies. 4 waves x 16-query tiles per block.
// ---------------------------------------------------------------------------
__global__ __launch_bounds__(128) void attn_kernel(
    const __bf16* __restrict__ Qb, const __bf16* __restrict__ Kb,
    const __bf16* __restrict__ VbT, __bf16* __restrict__ Yb)
{
  __shared__ __align__(16) __bf16 Ks[2][32][72];     // [key][d]
  __shared__ __align__(16) __bf16 Vt[2][64][40];     // [d][key]
  __shared__ __align__(16) __bf16 Ps[4][16][32];     // per-wave P tile

  const int qb = blockIdx.x, hh = blockIdx.y, bb = blockIdx.z;
  const int tid = threadIdx.x, lane = tid & 31, wv = tid >> 5;
  const int lm = lane & 15, hf = lane >> 4;
  const size_t off = (size_t)(bb * 16 + hh) * 2048 * 64;   // same for K and V^T
  const int q0 = qb * 64 + wv * 16;
  const unsigned long long kbase = (unsigned long long)Kb;
  const unsigned long long vbase = (unsigned long long)VbT;

  auto issue_kv = [&](int kb, int bf) {
    const int key0 = kb * 32;
#pragma unroll
    for (int i = 0; i < 2; ++i) {                     // K: 256 chunks
      int ci = tid + i * 128;
      int ky = ci >> 3, c8 = (ci & 7) << 3;
      async_b128((unsigned)(size_t)&Ks[bf][ky][c8],
                 (unsigned)((off + (size_t)(key0 + ky) * 64 + c8) * 2), kbase);
    }
#pragma unroll
    for (int i = 0; i < 2; ++i) {                     // V^T: 256 chunks
      int ci = tid + i * 128;
      int dd = ci >> 2, c8 = (ci & 3) << 3;
      async_b128((unsigned)(size_t)&Vt[bf][dd][c8],
                 (unsigned)((off + (size_t)dd * 2048 + key0 + c8) * 2), vbase);
    }
  };

  // preload Q as two A fragments (d 0..31, 32..63)
  const __bf16* qrow = Qb + off + (size_t)(q0 + lm) * 64;
  ABFrag a0, a1;
  a0.q[0] = *(const u32x4*)(qrow + hf * 16);
  a0.q[1] = *(const u32x4*)(qrow + hf * 16 + 8);
  a1.q[0] = *(const u32x4*)(qrow + 32 + hf * 16);
  a1.q[1] = *(const u32x4*)(qrow + 32 + hf * 16 + 8);

  float mrow[8], lrow[8];
  CFrag y[4];
#pragma unroll
  for (int r = 0; r < 8; ++r) { mrow[r] = -1e30f; lrow[r] = 0.0f; }
#pragma unroll
  for (int j = 0; j < 4; ++j)
#pragma unroll
    for (int r = 0; r < 8; ++r) y[j].e[r] = 0.0f;

  const int nkb = (qb + 1) * 2;
  issue_kv(0, 0);
  for (int kb = 0; kb < nkb; ++kb) {
    const int bf = kb & 1;
    const int key0 = kb * 32;
    if (kb + 1 < nkb) {
      issue_kv(kb + 1, bf ^ 1);
      asm volatile("s_wait_asynccnt 0x4" ::: "memory");
    } else {
      asm volatile("s_wait_asynccnt 0x0" ::: "memory");
    }
    __syncthreads();

    if (key0 <= q0 + 15) {               // wave-uniform causal guard
      CFrag s0, s1, zf;
#pragma unroll
      for (int r = 0; r < 8; ++r) zf.e[r] = 0.0f;
      ABFrag b00, b01, b10, b11;
      b00.q[0] = *(const u32x4*)&Ks[bf][lm][hf * 16];
      b00.q[1] = *(const u32x4*)&Ks[bf][lm][hf * 16 + 8];
      b01.q[0] = *(const u32x4*)&Ks[bf][lm][32 + hf * 16];
      b01.q[1] = *(const u32x4*)&Ks[bf][lm][32 + hf * 16 + 8];
      b10.q[0] = *(const u32x4*)&Ks[bf][16 + lm][hf * 16];
      b10.q[1] = *(const u32x4*)&Ks[bf][16 + lm][hf * 16 + 8];
      b11.q[0] = *(const u32x4*)&Ks[bf][16 + lm][32 + hf * 16];
      b11.q[1] = *(const u32x4*)&Ks[bf][16 + lm][32 + hf * 16 + 8];
      s0.v = WMMA_BF16(a0, b00, zf);
      s0.v = WMMA_BF16(a1, b01, s0);
      s1.v = WMMA_BF16(a0, b10, zf);
      s1.v = WMMA_BF16(a1, b11, s1);

      // online softmax: row stats live in lane halves matching C layout
#pragma unroll
      for (int r = 0; r < 8; ++r) {
        int mq  = q0 + r + hf * 8;
        int kn0 = key0 + lm, kn1 = kn0 + 16;
        float sl0 = (kn0 <= mq) ? s0.e[r] * 0.125f : -1e30f;
        float sl1 = (kn1 <= mq) ? s1.e[r] * 0.125f : -1e30f;
        float rm = fmaxf(sl0, sl1);
#pragma unroll
        for (int m = 1; m < 16; m <<= 1) rm = fmaxf(rm, __shfl_xor(rm, m, 32));
        float mnew  = fmaxf(mrow[r], rm);
        float alpha = exp2f((mrow[r] - mnew) * LOG2E);
        float p0 = exp2f((sl0 - mnew) * LOG2E);
        float p1 = exp2f((sl1 - mnew) * LOG2E);
        float rs = p0 + p1;
#pragma unroll
        for (int m = 1; m < 16; m <<= 1) rs += __shfl_xor(rs, m, 32);
        lrow[r] = lrow[r] * alpha + rs;
        mrow[r] = mnew;
#pragma unroll
        for (int j = 0; j < 4; ++j) y[j].e[r] *= alpha;
        Ps[wv][r + hf * 8][lm]      = (__bf16)p0;
        Ps[wv][r + hf * 8][16 + lm] = (__bf16)p1;
      }

      ABFrag pf;
      pf.q[0] = *(const u32x4*)&Ps[wv][lm][hf * 16];
      pf.q[1] = *(const u32x4*)&Ps[wv][lm][hf * 16 + 8];
#pragma unroll
      for (int j = 0; j < 4; ++j) {
        ABFrag vf;
        vf.q[0] = *(const u32x4*)&Vt[bf][j * 16 + lm][hf * 16];
        vf.q[1] = *(const u32x4*)&Vt[bf][j * 16 + lm][hf * 16 + 8];
        y[j].v = WMMA_BF16(pf, vf, y[j]);
      }
    }
    __syncthreads();
  }

#pragma unroll
  for (int r = 0; r < 8; ++r) {
    float inv = 1.0f / lrow[r];
    int row = q0 + r + hf * 8;
#pragma unroll
    for (int j = 0; j < 4; ++j) {
      Yb[((size_t)(bb * 2048 + row)) * 1024 + hh * 64 + j * 16 + lm] =
          (__bf16)(y[j].e[r] * inv);
    }
  }
}

// ---------------------------------------------------------------------------
// Phase 3: out = Yb @ WpbT^T + b_proj (fp32 out). Same pipelined tiling.
// ---------------------------------------------------------------------------
__global__ __launch_bounds__(128) void proj_gemm_kernel(
    const __bf16* __restrict__ Yb, const __bf16* __restrict__ WpbT,
    const float* __restrict__ bias, float* __restrict__ Out)
{
  __shared__ __align__(16) __bf16 As[2][128][40];
  __shared__ __align__(16) __bf16 Bs[2][64][40];

  const int row0 = blockIdx.x * 128;
  const int col0 = blockIdx.y * 64;
  const int tid  = threadIdx.x;
  const int lane = tid & 31, wv = tid >> 5;
  const int lm = lane & 15, hf = lane >> 4;
  const unsigned long long abase = (unsigned long long)Yb;
  const unsigned long long bbase = (unsigned long long)WpbT;

  auto issue_tile = [&](int k0, int bf) {
#pragma unroll
    for (int i = 0; i < 4; ++i) {
      int ci = tid + i * 128;
      int row = ci >> 2, c8 = (ci & 3) << 3;
      async_b128((unsigned)(size_t)&As[bf][row][c8],
                 (unsigned)(((row0 + row) * 1024 + k0 + c8) * 2), abase);
    }
#pragma unroll
    for (int i = 0; i < 2; ++i) {
      int ci = tid + i * 128;
      int nr = ci >> 2, c8 = (ci & 3) << 3;
      async_b128((unsigned)(size_t)&Bs[bf][nr][c8],
                 (unsigned)(((col0 + nr) * 1024 + k0 + c8) * 2), bbase);
    }
  };

  CFrag acc[2][4];
#pragma unroll
  for (int u = 0; u < 2; ++u)
#pragma unroll
    for (int j = 0; j < 4; ++j)
#pragma unroll
      for (int r = 0; r < 8; ++r) acc[u][j].e[r] = 0.0f;

  issue_tile(0, 0);
  for (int kt = 0; kt < 32; ++kt) {
    const int bf = kt & 1;
    if (kt + 1 < 32) {
      issue_tile((kt + 1) * 32, bf ^ 1);
      asm volatile("s_wait_asynccnt 0x6" ::: "memory");
    } else {
      asm volatile("s_wait_asynccnt 0x0" ::: "memory");
    }
    __syncthreads();

    ABFrag a0, a1;
    a0.q[0] = *(const u32x4*)&As[bf][wv * 32 + lm][hf * 8];
    a0.q[1] = *(const u32x4*)&As[bf][wv * 32 + lm][16 + hf * 8];
    a1.q[0] = *(const u32x4*)&As[bf][wv * 32 + 16 + lm][hf * 8];
    a1.q[1] = *(const u32x4*)&As[bf][wv * 32 + 16 + lm][16 + hf * 8];
#pragma unroll
    for (int j = 0; j < 4; ++j) {
      ABFrag b;
      b.q[0] = *(const u32x4*)&Bs[bf][j * 16 + lm][hf * 8];
      b.q[1] = *(const u32x4*)&Bs[bf][j * 16 + lm][16 + hf * 8];
      acc[0][j].v = WMMA_BF16(a0, b, acc[0][j]);
      acc[1][j].v = WMMA_BF16(a1, b, acc[1][j]);
    }
    __syncthreads();
  }

#pragma unroll
  for (int j = 0; j < 4; ++j) {
    int c = col0 + j * 16 + lm;
    float bv = bias[c];
#pragma unroll
    for (int u = 0; u < 2; ++u)
#pragma unroll
      for (int r = 0; r < 8; ++r) {
        int rr = row0 + wv * 32 + u * 16 + r + hf * 8;
        Out[(size_t)rr * 1024 + c] = acc[u][j].e[r] + bv;
      }
  }
}

// ---------------------------------------------------------------------------
extern "C" void kernel_launch(void* const* d_in, const int* in_sizes, int n_in,
                              void* d_out, int out_size, void* d_ws, size_t ws_size,
                              hipStream_t stream) {
  const float* X     = (const float*)d_in[0];
  const float* Wattn = (const float*)d_in[1];
  const float* battn = (const float*)d_in[2];
  const float* Wproj = (const float*)d_in[3];
  const float* bproj = (const float*)d_in[4];
  float* Out = (float*)d_out;

  // workspace layout (48 MB total):
  char* ws = (char*)d_ws;
  __bf16* Xb   = (__bf16*)(ws);                               // 8 MB  [4096,1024]
  __bf16* WabT = (__bf16*)(ws + (size_t) 8 * 1024 * 1024);    // 6 MB  [3072,1024]
  __bf16* WpbT = (__bf16*)(ws + (size_t)14 * 1024 * 1024);    // 2 MB  [1024,1024]
  __bf16* Qb   = (__bf16*)(ws + (size_t)16 * 1024 * 1024);    // 8 MB  [B,H,T,D]
  __bf16* Kb   = (__bf16*)(ws + (size_t)24 * 1024 * 1024);    // 8 MB  [B,H,T,D]
  __bf16* VbT  = (__bf16*)(ws + (size_t)32 * 1024 * 1024);    // 8 MB  [B,H,D,T]
  __bf16* Yb   = (__bf16*)(ws + (size_t)40 * 1024 * 1024);    // 8 MB  [B*T,C]

  cvt_kernel <<<4096, 256, 0, stream>>>(X, Xb, 4096 * 1024);
  cvtT_kernel<<<12288, 256, 0, stream>>>(Wattn, WabT, 1024, 3072);
  cvtT_kernel<<<4096, 256, 0, stream>>>(Wproj, WpbT, 1024, 1024);

  qkv_gemm_kernel<<<dim3(32, 48), 128, 0, stream>>>(Xb, WabT, battn, Qb, Kb, VbT);
  attn_kernel    <<<dim3(32, 16, 2), 128, 0, stream>>>(Qb, Kb, VbT, Yb);
  proj_gemm_kernel<<<dim3(32, 16), 128, 0, stream>>>(Yb, WpbT, bproj, Out);
}
